// LocalNormPool_22892175688204
// MI455X (gfx1250) — matrix operand order repaired and checked by reading, not compiled, not run on hardware
//
#include <hip/hip_runtime.h>
#include <hip/hip_bf16.h>
#include <math.h>

// B=16, G=2048, C=128, K=16
#define Bn 16
#define Gn 2048
#define Cn 128
#define Kn 16

typedef __attribute__((ext_vector_type(16))) _Float16 v16h;
typedef __attribute__((ext_vector_type(8)))  float    v8f;
typedef __attribute__((ext_vector_type(4)))  _Float16 h4;

// GCC-style int4 vector matching the async-LDS builtin's parameter type.
typedef int lnp_v4i __attribute__((vector_size(16)));
typedef __attribute__((address_space(1))) lnp_v4i* lnp_gp1;
typedef __attribute__((address_space(3))) lnp_v4i* lnp_lp3;

// ---------------------------------------------------------------------------
// CDNA5 async global->LDS copy (ASYNCcnt-tracked). Falls back to a plain
// vectorized copy if the builtin is not exposed by this toolchain.
// ---------------------------------------------------------------------------
__device__ inline void lnp_async_copy_b128(const _Float16* g, _Float16* l) {
#if __has_builtin(__builtin_amdgcn_global_load_async_to_lds_b128)
  __builtin_amdgcn_global_load_async_to_lds_b128(
      (lnp_gp1)(uintptr_t)g, (lnp_lp3)(uint32_t)(uintptr_t)l, 0, 0);
#else
  *reinterpret_cast<float4*>(l) = *reinterpret_cast<const float4*>(g);
#endif
}

__device__ inline void lnp_wait_async() {
#if __has_builtin(__builtin_amdgcn_global_load_async_to_lds_b128)
#if __has_builtin(__builtin_amdgcn_s_wait_asynccnt)
  __builtin_amdgcn_s_wait_asynccnt(0);
#else
  asm volatile("s_wait_asynccnt 0x0" ::: "memory");
#endif
#endif
}

// ---------------------------------------------------------------------------
// Kernel 1: transpose + f32->f16 convert the two 128x128 weight matrices so
// the WMMA B-operand (K-major per output column) loads as contiguous b128s.
// ---------------------------------------------------------------------------
__global__ void lnp_convert_weights(const float* __restrict__ W1,
                                    const float* __restrict__ W2,
                                    _Float16* __restrict__ W1t,
                                    _Float16* __restrict__ W2t) {
  const int i = blockIdx.x * blockDim.x + threadIdx.x;  // over 128*128
  const int k = i >> 7;
  const int n = i & 127;
  W1t[n * Cn + k] = (_Float16)W1[i];
  W2t[n * Cn + k] = (_Float16)W2[i];
}

// ---------------------------------------------------------------------------
// Kernel 2: fused KNN (exact top-16 by squared distance, low-index tiebreak,
// matching jax.lax.top_k) + neighbor-gather + L2 pool over K.
// Block = 256 threads = 8 waves; each wave owns one query row.
// Dynamic LDS: xyz of batch (2048*3 f32 = 24KB) + 8 distance rows (64KB).
// ---------------------------------------------------------------------------
__global__ __launch_bounds__(256) void lnp_knn_pool(
    const float* __restrict__ xyz, const float* __restrict__ feat,
    _Float16* __restrict__ pooled) {
  extern __shared__ float smemf[];
  float* xyzs = smemf;  // [Gn*3]
  const int tid  = threadIdx.x;
  const int wave = tid >> 5;
  const int lane = tid & 31;
  const int b      = blockIdx.x >> 8;   // 256 row-blocks per batch
  const int rowblk = blockIdx.x & 255;
  const int g      = rowblk * 8 + wave;
  float* drow = smemf + Gn * 3 + wave * Gn;  // per-wave distance row

  for (int i = tid; i < Gn * 3; i += 256)
    xyzs[i] = xyz[b * (Gn * 3) + i];
  __syncthreads();

  const float px = xyzs[g * 3 + 0];
  const float py = xyzs[g * 3 + 1];
  const float pz = xyzs[g * 3 + 2];

  // Each lane owns candidates c = lane + 32*j (same-lane LDS slots only).
  for (int c = lane; c < Gn; c += 32) {
    const float dx = px - xyzs[c * 3 + 0];
    const float dy = py - xyzs[c * 3 + 1];
    const float dz = pz - xyzs[c * 3 + 2];
    drow[c] = dx * dx + dy * dy + dz * dz;
  }

  // 16 argmin rounds; key = (f32 bits << 32) | idx  (d2 >= 0 so bit-monotone;
  // min key breaks ties toward the smaller index, like top_k).
  int sel[Kn];
#pragma unroll
  for (int r = 0; r < Kn; ++r) {
    unsigned long long best = ~0ull;
    for (int c = lane; c < Gn; c += 32) {
      const unsigned long long key =
          ((unsigned long long)__float_as_uint(drow[c]) << 32) | (unsigned)c;
      best = key < best ? key : best;
    }
#pragma unroll
    for (int off = 16; off > 0; off >>= 1) {
      const unsigned long long o = __shfl_xor(best, off, 32);
      best = o < best ? o : best;
    }
    const int widx = (int)(unsigned)best;
    sel[r] = widx;
    if ((widx & 31) == lane) drow[widx] = __builtin_inff();  // mask winner
  }

  // Gather the 16 neighbor rows, accumulate squares, sqrt, emit f16 pooled.
  float a0 = 0.f, a1 = 0.f, a2 = 0.f, a3 = 0.f;
  const float* fb = feat + (size_t)b * Gn * Cn;
#pragma unroll
  for (int r = 0; r < Kn; ++r) {
    if (r + 1 < Kn)
      __builtin_prefetch(fb + (size_t)sel[r + 1] * Cn + lane * 4, 0, 0);
    const float4 v =
        *reinterpret_cast<const float4*>(fb + (size_t)sel[r] * Cn + lane * 4);
    a0 += v.x * v.x; a1 += v.y * v.y; a2 += v.z * v.z; a3 += v.w * v.w;
  }
  h4 hv;
  hv[0] = (_Float16)sqrtf(a0);
  hv[1] = (_Float16)sqrtf(a1);
  hv[2] = (_Float16)sqrtf(a2);
  hv[3] = (_Float16)sqrtf(a3);
  *reinterpret_cast<h4*>(pooled + ((size_t)(b * Gn + g)) * Cn + lane * 4) = hv;
}

// ---------------------------------------------------------------------------
// WMMA fragment loaders (wave32), contiguous k order used by both A and B.
// A (16x32 f16): lane = (m, hi); elems 0..15 -> K = hi*16+{0..15}.
// B (32x16 f16): lane = (n, hi); elems 0..15 -> K = hi*16+{0..15}
//                from K-major (transposed) weights.
// ---------------------------------------------------------------------------
__device__ inline v16h lnp_load_afrag(const _Float16* __restrict__ base,
                                      int row, int kbase, int hi) {
  const _Float16* p = base + row * Cn + kbase + hi * 16;
  v16h a;
#pragma unroll
  for (int e = 0; e < 8; ++e) a[e] = p[e];
#pragma unroll
  for (int e = 0; e < 8; ++e) a[8 + e] = p[8 + e];
  return a;
}

__device__ inline v16h lnp_load_bfrag(const _Float16* __restrict__ wt,
                                      int ncol, int kbase, int hi) {
  const _Float16* p = wt + ncol * Cn + kbase + hi * 16;
  v16h bf;
#pragma unroll
  for (int e = 0; e < 16; ++e) bf[e] = p[e];
  return bf;
}

// ---------------------------------------------------------------------------
// Kernel 3: fused MLP: out = GELU(pooled@W1 + b1)@W2 + b2.
// 128 threads = 4 waves; each wave owns a 16-row M tile, computes all 8
// N-tiles; K=128 -> 4 chunks of 32 -> 32 WMMAs per GEMM, 64 per wave total.
// Both weight matrices staged once per block into LDS (async global->LDS on
// CDNA5), all 8 B fragments of a K-chunk preloaded before the WMMA chain so
// matrix ops are not serialized on s_wait_loadcnt 0.
// Dynamic LDS: 64KB weights + 16KB h tiles = 80KB.
// ---------------------------------------------------------------------------
__global__ __launch_bounds__(128) void lnp_mlp(
    const _Float16* __restrict__ pooled, const _Float16* __restrict__ Wt,
    const float* __restrict__ b1, const float* __restrict__ b2,
    float* __restrict__ out) {
  extern __shared__ _Float16 smemh[];  // [2*Cn*Cn weights][4*16*Cn h tiles]
  const int tid  = threadIdx.x;
  const int wave = tid >> 5;
  const int lane = tid & 31;
  const int m    = lane & 15;
  const int hi   = lane >> 4;
  const int rowbase = (blockIdx.x * 4 + wave) * 16;

  // Stage W1t|W2t (contiguous, 2*128*128 halves = 64KB) into LDS.
  for (int i = tid * 16; i < 2 * Cn * Cn; i += 128 * 16)
    lnp_async_copy_b128(Wt + i, smemh + i);
  lnp_wait_async();
  __syncthreads();

  const _Float16* wl1 = smemh;
  const _Float16* wl2 = smemh + Cn * Cn;
  _Float16* hs = smemh + 2 * Cn * Cn;

  const _Float16* Abase = pooled + (size_t)rowbase * Cn;

  // ---- GEMM1: h = pooled @ W1 ----
  v8f acc[8] = {};
#pragma unroll
  for (int kc = 0; kc < 4; ++kc) {
    v16h bbv[8];
#pragma unroll
    for (int nt = 0; nt < 8; ++nt)
      bbv[nt] = lnp_load_bfrag(wl1, nt * 16 + m, kc * 32, hi);
    const v16h a = lnp_load_afrag(Abase, m, kc * 32, hi);
#pragma unroll
    for (int nt = 0; nt < 8; ++nt)
      acc[nt] = __builtin_amdgcn_wmma_f32_16x16x32_f16(
          false, a, false, bbv[nt], (short)0, acc[nt], false, false);
  }

  // ---- bias + exact GELU, spill h tile to LDS as f16 (row-major 16x128) ----
  _Float16* hrow = hs + wave * 16 * Cn;
#pragma unroll
  for (int nt = 0; nt < 8; ++nt) {
    const int n = nt * 16 + m;
    const float bn = b1[n];
#pragma unroll
    for (int v = 0; v < 8; ++v) {
      const int M = v + 8 * hi;  // C/D layout: VGPR v, lanes 16-31 -> M=v+8
      const float x = acc[nt][v] + bn;
      const float hval = 0.5f * x * (1.0f + erff(x * 0.70710678118654752f));
      hrow[M * Cn + n] = (_Float16)hval;
    }
  }
  __syncthreads();

  // ---- GEMM2: out = h @ W2 ----
  v8f acc2[8] = {};
#pragma unroll
  for (int kc = 0; kc < 4; ++kc) {
    v16h bbv[8];
#pragma unroll
    for (int nt = 0; nt < 8; ++nt)
      bbv[nt] = lnp_load_bfrag(wl2, nt * 16 + m, kc * 32, hi);
    const v16h a = lnp_load_afrag(hrow, m, kc * 32, hi);
#pragma unroll
    for (int nt = 0; nt < 8; ++nt)
      acc2[nt] = __builtin_amdgcn_wmma_f32_16x16x32_f16(
          false, a, false, bbv[nt], (short)0, acc2[nt], false, false);
  }

  float* orow = out + (size_t)rowbase * Cn;
#pragma unroll
  for (int nt = 0; nt < 8; ++nt) {
    const int n = nt * 16 + m;
    const float bn = b2[n];
#pragma unroll
    for (int v = 0; v < 8; ++v) {
      const int M = v + 8 * hi;
      orow[M * Cn + n] = acc2[nt][v] + bn;
    }
  }
}

// ---------------------------------------------------------------------------
extern "C" void kernel_launch(void* const* d_in, const int* in_sizes, int n_in,
                              void* d_out, int out_size, void* d_ws,
                              size_t ws_size, hipStream_t stream) {
  (void)in_sizes; (void)n_in; (void)out_size; (void)ws_size;
  const float* xyz  = (const float*)d_in[0];
  const float* feat = (const float*)d_in[1];
  const float* W1   = (const float*)d_in[2];
  const float* b1   = (const float*)d_in[3];
  const float* W2   = (const float*)d_in[4];
  const float* b2   = (const float*)d_in[5];
  float* out = (float*)d_out;

  // Workspace layout: W1t f16 (32KB) | W2t f16 (32KB, contiguous) | pooled f16
  _Float16* W1t    = (_Float16*)d_ws;
  _Float16* W2t    = W1t + Cn * Cn;
  _Float16* pooled = W2t + Cn * Cn;

  lnp_convert_weights<<<(Cn * Cn) / 256, 256, 0, stream>>>(W1, W2, W1t, W2t);

  const size_t knn_lds = (size_t)(Gn * 3 + 8 * Gn) * sizeof(float);  // 88KB
  lnp_knn_pool<<<Bn * (Gn / 8), 256, knn_lds, stream>>>(xyz, feat, pooled);

  const size_t mlp_lds = (size_t)(2 * Cn * Cn + 4 * 16 * Cn) * sizeof(_Float16);  // 80KB
  lnp_mlp<<<(Bn * Gn) / 64, 128, mlp_lds, stream>>>(pooled, W1t, b1, b2, out);
}
